// MoE_85950885528518
// MI455X (gfx1250) — compile-verified, matrix operand order checked by reading, not run on hardware
//
#include <hip/hip_runtime.h>
#include <hip/hip_bf16.h>

typedef __attribute__((ext_vector_type(16))) _Float16 v16h;
typedef __attribute__((ext_vector_type(8)))  _Float16 v8h;
typedef __attribute__((ext_vector_type(8)))  float    v8f;

#define N_TOK 32768
#define C_DIM 256
#define E_EXP 16

// ---------------- stage 0: conversions ----------------

__global__ void k_zero(float* __restrict__ p, int n) {
    int i = blockIdx.x * blockDim.x + threadIdx.x;
    if (i < n) p[i] = 0.0f;
}

__global__ void k_cvt_inputs(const float* __restrict__ x, _Float16* __restrict__ xh, int n) {
    int i = blockIdx.x * blockDim.x + threadIdx.x;
    if (i < n) xh[i] = (_Float16)x[i];
}

// transpose+convert share_W (m==0) and expert_W (m=1..16) to f16, N-major:
// Wt[c*C + d] = W[d*C + c]  so B-fragment loads are contiguous in K.
__global__ void k_cvt_weights(const float* __restrict__ sW, const float* __restrict__ eW,
                              _Float16* __restrict__ sWt, _Float16* __restrict__ eWt) {
    int i = blockIdx.x * blockDim.x + threadIdx.x;
    int total = (1 + E_EXP) * C_DIM * C_DIM;
    if (i >= total) return;
    int m  = i / (C_DIM * C_DIM);
    int cd = i % (C_DIM * C_DIM);
    int c = cd / C_DIM, d = cd % C_DIM;
    if (m == 0) sWt[c * C_DIM + d] = (_Float16)sW[d * C_DIM + c];
    else {
        int e = m - 1;
        eWt[e * C_DIM * C_DIM + c * C_DIM + d] =
            (_Float16)eW[e * C_DIM * C_DIM + d * C_DIM + c];
    }
}

// ---------------- stage 1: router (LN + gate + softmax + dynamic top-k) ----------------

__global__ void k_router(const float* __restrict__ conv, const float* __restrict__ demb,
                         const float* __restrict__ gW,   const float* __restrict__ gb,
                         const float* __restrict__ gamma, const float* __restrict__ beta,
                         float* __restrict__ combine, float* __restrict__ bal /*[32]*/) {
    __shared__ float routeLDS[8][C_DIM];
    __shared__ float ftAcc[E_EXP];
    __shared__ float fpAcc[E_EXP];
    int tid  = threadIdx.x;
    int lane = tid & 31;
    int wid  = tid >> 5;          // 8 waves per block, one token per wave per iter
    if (tid < E_EXP)               ftAcc[tid] = 0.0f;
    else if (tid < 2 * E_EXP)      fpAcc[tid - E_EXP] = 0.0f;
    __syncthreads();

    int slot   = blockIdx.x * 8 + wid;
    int nSlots = gridDim.x * 8;   // grid chosen so N_TOK % nSlots == 0 (uniform trips)
    for (int n = slot; n < N_TOK; n += nSlots) {
        // --- LayerNorm over C=256, 8 elems/lane ---
        float vals[8], s = 0.0f, ss = 0.0f;
        #pragma unroll
        for (int i = 0; i < 8; i++) {
            float v = conv[n * C_DIM + lane + 32 * i];
            vals[i] = v; s += v; ss += v * v;
        }
        #pragma unroll
        for (int m = 16; m > 0; m >>= 1) {
            s  += __shfl_xor(s,  m, 32);
            ss += __shfl_xor(ss, m, 32);
        }
        float mu   = s  * (1.0f / C_DIM);
        float var  = ss * (1.0f / C_DIM) - mu * mu;
        float rstd = rsqrtf(var + 1e-5f);
        #pragma unroll
        for (int i = 0; i < 8; i++) {
            int d = lane + 32 * i;
            routeLDS[wid][d] = (vals[i] - mu) * rstd * gamma[d] + beta[d] + demb[d];
        }
        __syncthreads();

        // --- gate logits: lane (&15) owns expert e; both halves replicate ---
        int e = lane & 15;
        float logit = gb[e];
        for (int d = 0; d < C_DIM; d++)
            logit += routeLDS[wid][d] * gW[d * E_EXP + e];

        // --- softmax over 16 experts (width-16 reductions) ---
        float mx = logit;
        #pragma unroll
        for (int m = 8; m > 0; m >>= 1) mx = fmaxf(mx, __shfl_xor(mx, m, 16));
        float p = __expf(logit - mx);
        float sum = p;
        #pragma unroll
        for (int m = 8; m > 0; m >>= 1) sum += __shfl_xor(sum, m, 16);
        float w = p / sum;

        // --- entropy -> dynamic k ---
        float ent = -w * __logf(w + 1e-12f);
        #pragma unroll
        for (int m = 8; m > 0; m >>= 1) ent += __shfl_xor(ent, m, 16);
        int k = (int)ceilf(1.0f + ent * (15.0f / 2.7725887f));   // 15/ln(16)
        k = min(max(k, 1), 16);

        // --- stable descending rank (matches argsort(-w) tie-break by index) ---
        int rank = 0;
        #pragma unroll
        for (int j = 0; j < 16; j++) {
            float wj = __shfl(w, (lane & 16) + j, 32);
            rank += (wj > w) || (wj == w && j < e);
        }
        bool  sel = rank < k;
        float cmb = sel ? w : 0.0f;

        if (lane < 16) {
            combine[n * E_EXP + e] = cmb;
            atomicAdd(&ftAcc[e], sel ? 1.0f : 0.0f);
            atomicAdd(&fpAcc[e], w);
        }
        __syncthreads();
    }
    if (tid < E_EXP) {
        atomicAdd(&bal[tid],         ftAcc[tid]);
        atomicAdd(&bal[E_EXP + tid], fpAcc[tid]);
    }
}

// ---------------- stage 2: fused shared + expert GEMM with combine ----------------
// block = 512 threads = 16 waves. Block owns 256 tokens x one 16-column stripe.
// Wave w owns tokens [blockTok + 16w, +16); A fragments register-resident (8 x v16h),
// reused across all 17 B matrices. Each 8KB B tile staged in LDS once per block
// (double-buffered), reused by all 16 waves -> B arithmetic intensity 256 FLOP/byte.

__global__ void k_moe_gemm(const _Float16* __restrict__ Ah,  const _Float16* __restrict__ SWt,
                           const _Float16* __restrict__ EWt, const float* __restrict__ sb,
                           const float* __restrict__ eb,     const float* __restrict__ combine,
                           float* __restrict__ out) {
    __shared__ _Float16 Bbuf[2][16 * C_DIM];   // 2 x 8 KB double buffer
    int tid  = threadIdx.x;
    int lane = tid & 31;
    int wid  = tid >> 5;                       // 0..15
    int tokGroup = blockIdx.x >> 4;            // 128 groups of 256 tokens
    int colGroup = blockIdx.x & 15;            // 16 column stripes
    int blockTok = tokGroup * 256;
    int myTok    = blockTok + wid * 16;        // this wave's 16 tokens
    int colBase  = colGroup * 16;
    int r  = lane & 15;                        // M for A, N for B, column for C/D
    int hi = lane >> 4;

    // ---- hoist all 8 A fragments (K chunks of 32) straight from global ----
    // lane r holds row M=r halves; K = 32*kc + 16*(j>=8) + (j&7) + 8*hi
    v16h afr[8];
    #pragma unroll
    for (int kc = 0; kc < 8; kc++) {
        const _Float16* arow = Ah + (myTok + r) * C_DIM + kc * 32 + hi * 8;
        v8h lo = *(const v8h*)(arow);
        v8h hh = *(const v8h*)(arow + 16);
        #pragma unroll
        for (int j = 0; j < 8; j++) { afr[kc][j] = lo[j]; afr[kc][8 + j] = hh[j]; }
    }

    // B-tile staging: 16 rows x 256 halfs = 512 x 16B; one v8h per thread
    int srow = tid >> 5;                 // 0..15  (row within stripe)
    int scol = (tid & 31) * 8;           // 0..248 (half offset in K)
    {
        v8h b0 = *(const v8h*)(SWt + (colBase + srow) * C_DIM + scol);
        *(v8h*)(&Bbuf[0][srow * C_DIM + scol]) = b0;
    }
    __syncthreads();

    auto loadB = [&](const _Float16* Bl, int kc) -> v16h {
        const _Float16* brow = Bl + r * C_DIM + kc * 32 + hi * 8;
        v8h lo = *(const v8h*)(brow);
        v8h hh = *(const v8h*)(brow + 16);
        v16h b;
        #pragma unroll
        for (int j = 0; j < 8; j++) { b[j] = lo[j]; b[8 + j] = hh[j]; }
        return b;
    };

    v8f acc = {};
    for (int m = 0; m < 1 + E_EXP; m++) {
        // prefetch next matrix's B tile into registers (overlaps with WMMA below)
        v8h nreg;
        if (m < E_EXP)
            nreg = *(const v8h*)(EWt + m * C_DIM * C_DIM + (colBase + srow) * C_DIM + scol);

        // ---- 8 WMMAs over K=256, two independent accumulator chains ----
        const _Float16* Bl = &Bbuf[m & 1][0];
        v8f a0 = {}, a1 = {};
        #pragma unroll
        for (int kc = 0; kc < 8; kc += 2) {
            a0 = __builtin_amdgcn_wmma_f32_16x16x32_f16(false, afr[kc],     false,
                                                        loadB(Bl, kc),     (short)0, a0,
                                                        false, false);
            a1 = __builtin_amdgcn_wmma_f32_16x16x32_f16(false, afr[kc + 1], false,
                                                        loadB(Bl, kc + 1), (short)0, a1,
                                                        false, false);
        }

        if (m == 0) {           // shared expert: direct accumulate + bias
            float bias = sb[colBase + r];
            #pragma unroll
            for (int v = 0; v < 8; v++) acc[v] = a0[v] + a1[v] + bias;
        } else {                // expert m-1: per-row combine scaling (row M = v + 8*hi)
            int e = m - 1;
            float bias = eb[e * C_DIM + colBase + r];
            #pragma unroll
            for (int v = 0; v < 8; v++) {
                float cmb = combine[(myTok + v + 8 * hi) * E_EXP + e];
                acc[v] += cmb * (a0[v] + a1[v] + bias);
            }
        }

        // commit prefetched tile into the other LDS buffer
        if (m < E_EXP)
            *(v8h*)(&Bbuf[(m + 1) & 1][srow * C_DIM + scol]) = nreg;
        __syncthreads();
    }

    // store: lane column = colBase + r, rows myTok + v + 8*hi
    #pragma unroll
    for (int v = 0; v < 8; v++)
        out[(myTok + v + 8 * hi) * C_DIM + colBase + r] = acc[v];
}

// ---------------- stage 3: balance loss ----------------

__global__ void k_loss(const float* __restrict__ bal, float* __restrict__ out_loss) {
    if (threadIdx.x == 0) {
        float s = 0.0f;
        for (int e = 0; e < E_EXP; e++)
            s += (bal[e] * (1.0f / N_TOK)) * (bal[E_EXP + e] * (1.0f / N_TOK));
        // loss = mean(ft*fp) * E^2 = (s/16) * 256 = 16*s
        out_loss[0] = 16.0f * s;
    }
}

// ---------------- host launcher ----------------

extern "C" void kernel_launch(void* const* d_in, const int* in_sizes, int n_in,
                              void* d_out, int out_size, void* d_ws, size_t ws_size,
                              hipStream_t stream) {
    const float* inputs   = (const float*)d_in[0];
    const float* conv_out = (const float*)d_in[1];
    const float* demb     = (const float*)d_in[2];
    const float* share_W  = (const float*)d_in[3];
    const float* share_b  = (const float*)d_in[4];
    const float* gate_W   = (const float*)d_in[5];
    const float* gate_b   = (const float*)d_in[6];
    const float* expert_W = (const float*)d_in[7];
    const float* expert_b = (const float*)d_in[8];
    const float* ln_gamma = (const float*)d_in[9];
    const float* ln_beta  = (const float*)d_in[10];
    float* out = (float*)d_out;              // [N*C] moe_out+share_out, then [1] loss

    char* ws = (char*)d_ws;
    _Float16* Ah      = (_Float16*)(ws);                          // 16,777,216 B
    _Float16* SWt     = (_Float16*)(ws + 16777216);               //    131,072 B
    _Float16* EWt     = (_Float16*)(ws + 16908288);               //  2,097,152 B
    float*    combine = (float*)   (ws + 19005440);               //  2,097,152 B
    float*    bal     = (float*)   (ws + 21102592);               //        128 B

    // zero balance accumulators (ws is poisoned)
    k_zero<<<1, 32, 0, stream>>>(bal, 32);

    // f16 conversions
    k_cvt_inputs<<<(N_TOK * C_DIM + 255) / 256, 256, 0, stream>>>(inputs, Ah, N_TOK * C_DIM);
    k_cvt_weights<<<((1 + E_EXP) * C_DIM * C_DIM + 255) / 256, 256, 0, stream>>>(
        share_W, expert_W, SWt, EWt);

    // router: 512 blocks x 8 waves -> 4096 slots, 8 tokens each (uniform)
    k_router<<<512, 256, 0, stream>>>(conv_out, demb, gate_W, gate_b,
                                      ln_gamma, ln_beta, combine, bal);

    // fused GEMM: 128 token groups x 16 column stripes
    k_moe_gemm<<<(N_TOK / 256) * 16, 512, 0, stream>>>(Ah, SWt, EWt, share_b, expert_b,
                                                       combine, out);

    // loss scalar at out[N*C]
    k_loss<<<1, 32, 0, stream>>>(bal, out + N_TOK * C_DIM);
}
